// LogisticGaussianHMM_33054068310025
// MI455X (gfx1250) — matrix operand order repaired and checked by reading, not compiled, not run on hardware
//
#include <hip/hip_runtime.h>
#include <hip/hip_bf16.h>

typedef __attribute__((ext_vector_type(16))) _Float16 v16h;
typedef __attribute__((ext_vector_type(8)))  float    v8f;

#define T_DIM 512
#define B_DIM 64
#define S_DIM 32
#define D_DIM 32
#define A_DIM 8
#define EPSF  1e-6f

__device__ __forceinline__ float warp_max32(float v) {
  #pragma unroll
  for (int o = 16; o > 0; o >>= 1) v = fmaxf(v, __shfl_xor(v, o, 32));
  return v;
}
__device__ __forceinline__ float warp_sum32(float v) {
  #pragma unroll
  for (int o = 16; o > 0; o >>= 1) v += __shfl_xor(v, o, 32);
  return v;
}

// ---------------------------------------------------------------------------
// K0: per-state Cholesky-style factor L = tril(tril_off,-1)+diag(exp(log_sigma)),
// forward-substitution inverse Linv, v = Linv@mu, const term. 32 blocks x 32 thr.
// ---------------------------------------------------------------------------
__global__ void hmm_prep(const float* __restrict__ mu,
                         const float* __restrict__ log_sigma,
                         const float* __restrict__ tril,
                         _Float16* __restrict__ Mf16,
                         float* __restrict__ vvec,
                         float* __restrict__ cconst) {
  const int s = blockIdx.x;
  const int c = threadIdx.x;           // column of Linv this lane solves
  __shared__ float L[32][33];
  __shared__ float Li[32][33];
  #pragma unroll
  for (int i = 0; i < 32; ++i) {
    float v;
    if (i > c)       v = tril[s * 1024 + i * 32 + c];
    else if (i == c) v = expf(log_sigma[s * 32 + i]);
    else             v = 0.f;
    L[i][c] = v;
  }
  __syncthreads();
  float y[32];
  #pragma unroll
  for (int i = 0; i < 32; ++i) {
    float t = (i == c) ? 1.f : 0.f;
    #pragma unroll
    for (int j = 0; j < 32; ++j)
      if (j < i) t -= L[i][j] * y[j];
    y[i] = (i < c) ? 0.f : t / L[i][i];
  }
  #pragma unroll
  for (int i = 0; i < 32; ++i) Li[i][c] = y[i];
  __syncthreads();
  // lane c now handles row c of Linv
  float vv = 0.f;
  #pragma unroll
  for (int j = 0; j < 32; ++j) vv += Li[c][j] * mu[s * 32 + j];
  vvec[s * 32 + c] = vv;
  #pragma unroll
  for (int j = 0; j < 32; ++j)
    Mf16[s * 1024 + c * 32 + j] = (_Float16)Li[c][j];
  float tot = warp_sum32(log_sigma[s * 32 + c]);
  if (c == 0) cconst[s] = -16.f * 1.8378770664093453f - tot;  // -0.5*D*log(2pi) - sum(log_sigma)
}

// ---------------------------------------------------------------------------
// K1: logp_x[t,b,s] = cconst[s] - 0.5*||Linv[s]x - v[s]||^2 via WMMA f16.
// Y^T formulation: A = Linv rows (16 x 32), B = X_tile^T (32 x 16).
// 256 threads (8 waves), 16 rows/wave, 128 rows/block, 256 blocks.
// Linv (f16) staged through LDS in two 16-state passes (32 KB each).
// ---------------------------------------------------------------------------
__global__ void hmm_logp(const float* __restrict__ x,
                         const _Float16* __restrict__ Mf16,
                         const float* __restrict__ vvec,
                         const float* __restrict__ cconst,
                         float* __restrict__ logp) {
  __shared__ _Float16 Msh[16 * 1024];
  __shared__ float    vsh[32 * 32];
  __shared__ float    csh[32];
  const int tid  = threadIdx.x;
  const int wave = tid >> 5;
  const int lane = tid & 31;
  for (int i = tid; i < 1024; i += 256) vsh[i] = vvec[i];
  if (tid < 32) csh[tid] = cconst[tid];

  const int rbase = blockIdx.x * 128 + wave * 16;

  // B fragment: 32x16 f16, lane: n = lane&15, K-half = (lane<16)?0:16,
  // VGPR p holds K = kb+2p, kb+2p+1  (reused for every state)
  v16h bfrag;
  const int n  = lane & 15;
  const int kb = (lane < 16) ? 0 : 16;
  const float* xr = x + (rbase + n) * D_DIM + kb;
  #pragma unroll
  for (int p = 0; p < 8; ++p) {
    float2 f = *(const float2*)(xr + 2 * p);
    bfrag[2 * p]     = (_Float16)f.x;
    bfrag[2 * p + 1] = (_Float16)f.y;
  }

  const int m16  = lane & 15;
  const int kbb  = (lane < 16) ? 0 : 16;
  const int ioff = (lane < 16) ? 0 : 8;

  for (int pass = 0; pass < 2; ++pass) {
    __syncthreads();
    {  // stage 16 states of Linv f16 into LDS (32 KB)
      const unsigned int* src = (const unsigned int*)(Mf16 + pass * 16 * 1024);
      unsigned int* dst = (unsigned int*)Msh;
      for (int i = tid; i < 8192; i += 256) dst[i] = src[i];
    }
    __syncthreads();
    for (int sl = 0; sl < 16; ++sl) {
      const int s = pass * 16 + sl;
      const _Float16* Mb = Msh + sl * 1024;
      v16h a0, a1;
      // A fragment (16x32 f16): lanes 0-15/16-31 both hold rows 0-15,
      // K contiguous: VGPR p holds K = kbb+2p, kbb+2p+1; lane-half adds +16
      #pragma unroll
      for (int p = 0; p < 8; ++p) {
        const int kk = kbb + 2 * p;
        a0[2 * p]     = Mb[m16 * 32 + kk];
        a0[2 * p + 1] = Mb[m16 * 32 + kk + 1];
        a1[2 * p]     = Mb[(16 + m16) * 32 + kk];
        a1[2 * p + 1] = Mb[(16 + m16) * 32 + kk + 1];
      }
      v8f c0 = {};
      v8f c1 = {};
      c0 = __builtin_amdgcn_wmma_f32_16x16x32_f16(false, a0, false, bfrag,
                                                  (short)0, c0, false, false);
      c1 = __builtin_amdgcn_wmma_f32_16x16x32_f16(false, a1, false, bfrag,
                                                  (short)0, c1, false, false);
      // C layout: VGPR p -> row i = p (+8 for upper lane half); col = lane&15
      const float* vs = vsh + s * 32;
      float ssq = 0.f;
      #pragma unroll
      for (int p = 0; p < 8; ++p) {
        float d0 = c0[p] - vs[ioff + p];
        float d1 = c1[p] - vs[16 + ioff + p];
        ssq += d0 * d0 + d1 * d1;
      }
      ssq += __shfl_xor(ssq, 16, 32);  // combine the two i-half lane groups
      float lp = csh[s] - 0.5f * ssq;
      if (lane < 16) logp[(rbase + n) * S_DIM + s] = lp;
    }
  }
}

// ---------------------------------------------------------------------------
// K2: trans[t,b,j,k] = softmax_k( a[t,b,:]@W[:, j*32+k] + bias ).
// rows = (T-1)*B = 32704; 8 rows/block (1 row/wave), W+bias in LDS.
// ---------------------------------------------------------------------------
__global__ void hmm_trans(const float* __restrict__ a,
                          const float* __restrict__ W,
                          const float* __restrict__ bias,
                          float* __restrict__ trans) {
  __shared__ float Wsh[8 * 1024];
  __shared__ float bsh[1024];
  const int tid = threadIdx.x;
  for (int i = tid; i < 8192; i += 256) Wsh[i] = W[i];
  for (int i = tid; i < 1024; i += 256) bsh[i] = bias[i];
  __syncthreads();
  const int lane = tid & 31;
  const int row  = blockIdx.x * 8 + (tid >> 5);  // t*B+b, t<T-1
  float ar[A_DIM];
  #pragma unroll
  for (int q = 0; q < A_DIM; ++q) ar[q] = a[row * A_DIM + q];
  float* out = trans + (size_t)row * 1024;
  for (int j = 0; j < 32; ++j) {
    const int m = j * 32 + lane;
    float logit = bsh[m];
    #pragma unroll
    for (int q = 0; q < A_DIM; ++q) logit += ar[q] * Wsh[q * 1024 + m];
    float mx = warp_max32(logit);
    float e  = expf(logit - mx);
    float sm = warp_sum32(e);
    out[m] = e / sm;
  }
}

// ---------------------------------------------------------------------------
// K3: forward scan (alpha). One wave per batch b; lane = state k.
// ---------------------------------------------------------------------------
__global__ void hmm_forward(const float* __restrict__ logp,
                            const float* __restrict__ trans,
                            const float* __restrict__ z0,
                            float* __restrict__ alpha) {
  const int b = blockIdx.x;
  const int k = threadIdx.x;
  // pi0 = softmax(z0) (added in prob space, as in source)
  float z  = z0[k];
  float mz = warp_max32(z);
  float ez = expf(z - mz);
  float pz = ez / warp_sum32(ez);
  // alpha[0]
  float val = logp[b * S_DIM + k] + pz;
  float mx  = warp_max32(val);
  float e   = expf(val - mx);
  float ap  = e / warp_sum32(e);
  alpha[b * S_DIM + k] = ap;
  for (int t = 1; t < T_DIM; ++t) {
    const float* tr = trans + ((size_t)(t - 1) * B_DIM + b) * 1024;
    if (t + 1 < T_DIM)
      __builtin_prefetch(trans + ((size_t)t * B_DIM + b) * 1024, 0, 3);
    float s = 0.f;
    #pragma unroll
    for (int j = 0; j < 32; ++j)
      s += __shfl(ap, j, 32) * tr[j * 32 + k];
    float lz = logf(s + 32.f * EPSF);  // sum_j(tr*ap + EPS)
    float v  = logp[(t * B_DIM + b) * S_DIM + k] + lz;
    float m2 = warp_max32(v);
    float e2 = expf(v - m2);
    ap = e2 / warp_sum32(e2);
    alpha[(t * B_DIM + b) * S_DIM + k] = ap;
  }
}

// ---------------------------------------------------------------------------
// K4: backward scan (log_beta). One wave per batch b; lane = state index.
// ---------------------------------------------------------------------------
__global__ void hmm_backward(const float* __restrict__ logp,
                             const float* __restrict__ trans,
                             const float* __restrict__ mask,
                             float* __restrict__ lbeta) {
  const int b = blockIdx.x;
  const int j = threadIdx.x;   // lane carries both j (compute) and k (state)
  float lb = 0.f;
  lbeta[((T_DIM - 1) * B_DIM + b) * S_DIM + j] = 0.f;
  for (int t = T_DIM - 2; t >= 0; --t) {
    const float lpl = logp[((t + 1) * B_DIM + b) * S_DIM + j] + lb;  // lpx+lb at k=lane
    const float* tr = trans + ((size_t)t * B_DIM + b) * 1024 + j * 32;
    if (t > 0)
      __builtin_prefetch(trans + ((size_t)(t - 1) * B_DIM + b) * 1024 + j * 32, 0, 3);
    float w[32];
    float mx = -3.4e38f;
    #pragma unroll
    for (int k = 0; k < 32; ++k) {
      float term = __shfl(lpl, k, 32) + logf(tr[k] + EPSF);
      w[k] = term;
      mx = fmaxf(mx, term);
    }
    float ss = 0.f;
    #pragma unroll
    for (int k = 0; k < 32; ++k) ss += expf(w[k] - mx);
    float lbn = mx + logf(ss);
    float m   = mask[(t + 1) * B_DIM + b];
    lb = (m == 1.0f) ? lbn : 0.f;
    lbeta[((size_t)t * B_DIM + b) * S_DIM + j] = lb;
  }
}

// ---------------------------------------------------------------------------
// K5: q_z = softmax(log(alpha+EPS) + log_beta) over S. One wave per (t,b) row.
// ---------------------------------------------------------------------------
__global__ void hmm_qz(const float* __restrict__ alpha,
                       const float* __restrict__ lbeta,
                       float* __restrict__ qz) {
  const int row = blockIdx.x * 8 + (threadIdx.x >> 5);
  const int k   = threadIdx.x & 31;
  float v  = logf(alpha[row * S_DIM + k] + EPSF) + lbeta[row * S_DIM + k];
  float mx = warp_max32(v);
  float e  = expf(v - mx);
  qz[row * S_DIM + k] = e / warp_sum32(e);
}

// ---------------------------------------------------------------------------
// K6: q_zz = softmax over S*S of
//   log_alpha[t+1,b,j] + trans[t,b,j,k] + logp[t+1,b,k] + log_beta[t,b,k].
// One wave per (t,b) tile; lane = k; j-axis in registers.
// ---------------------------------------------------------------------------
__global__ void hmm_qzz(const float* __restrict__ alpha,
                        const float* __restrict__ lbeta,
                        const float* __restrict__ logp,
                        const float* __restrict__ trans,
                        float* __restrict__ qzz) {
  const int tile = blockIdx.x * 8 + (threadIdx.x >> 5);  // t*B+b, t<T-1
  const int k    = threadIdx.x & 31;
  const int t    = tile / B_DIM;
  const int b    = tile % B_DIM;
  const float la  = logf(alpha[((t + 1) * B_DIM + b) * S_DIM + k] + EPSF);  // j=k slot
  const float lpk = logp[((t + 1) * B_DIM + b) * S_DIM + k];
  const float lbk = lbeta[((size_t)t * B_DIM + b) * S_DIM + k];
  const float* tr = trans + (size_t)tile * 1024;
  float w[32];
  float mx = -3.4e38f;
  #pragma unroll
  for (int j = 0; j < 32; ++j) {
    float v = __shfl(la, j, 32) + tr[j * 32 + k] + lpk + lbk;
    w[j] = v;
    mx = fmaxf(mx, v);
  }
  mx = warp_max32(mx);
  float ss = 0.f;
  #pragma unroll
  for (int j = 0; j < 32; ++j) {
    w[j] = expf(w[j] - mx);
    ss += w[j];
  }
  ss = warp_sum32(ss);
  const float inv = 1.f / ss;
  float* out = qzz + (size_t)tile * 1024;
  #pragma unroll
  for (int j = 0; j < 32; ++j) out[j * 32 + k] = w[j] * inv;
}

// ---------------------------------------------------------------------------
extern "C" void kernel_launch(void* const* d_in, const int* in_sizes, int n_in,
                              void* d_out, int out_size, void* d_ws, size_t ws_size,
                              hipStream_t stream) {
  const float* x         = (const float*)d_in[0];  // [T,B,D]
  const float* a         = (const float*)d_in[1];  // [T,B,A]
  const float* mask      = (const float*)d_in[2];  // [T,B]
  const float* mu        = (const float*)d_in[3];  // [S,D]
  const float* log_sigma = (const float*)d_in[4];  // [S,D]
  const float* tril_off  = (const float*)d_in[5];  // [S,D,D]
  const float* w_trans   = (const float*)d_in[6];  // [A,S*S]
  const float* b_trans   = (const float*)d_in[7];  // [S*S]
  const float* z0        = (const float*)d_in[8];  // [S]

  // workspace layout (bytes, 256-aligned)
  char* ws = (char*)d_ws;
  _Float16* Mf16  = (_Float16*)ws;                       ws += 65536;        // [S,32,32] f16
  float*    vvec  = (float*)ws;                          ws += 4096;         // [S,32]
  float*    cconst= (float*)ws;                          ws += 256;          // [S]
  float*    logp  = (float*)ws;                          ws += (size_t)T_DIM * B_DIM * S_DIM * 4;        // [T,B,S]
  float*    alpha = (float*)ws;                          ws += (size_t)T_DIM * B_DIM * S_DIM * 4;        // [T,B,S]
  float*    lbeta = (float*)ws;                          ws += (size_t)T_DIM * B_DIM * S_DIM * 4;        // [T,B,S]
  float*    trans = (float*)ws;                                                                         // [T-1,B,S,S]

  float* qz  = (float*)d_out;                            // [T,B,S]
  float* qzz = (float*)d_out + (size_t)T_DIM * B_DIM * S_DIM;  // [T-1,B,S,S]

  hmm_prep<<<S_DIM, 32, 0, stream>>>(mu, log_sigma, tril_off, Mf16, vvec, cconst);
  hmm_logp<<<(T_DIM * B_DIM) / 128, 256, 0, stream>>>(x, Mf16, vvec, cconst, logp);
  hmm_trans<<<((T_DIM - 1) * B_DIM) / 8, 256, 0, stream>>>(a, w_trans, b_trans, trans);
  hmm_forward<<<B_DIM, 32, 0, stream>>>(logp, trans, z0, alpha);
  hmm_backward<<<B_DIM, 32, 0, stream>>>(logp, trans, mask, lbeta);
  hmm_qz<<<(T_DIM * B_DIM) / 8, 256, 0, stream>>>(alpha, lbeta, qz);
  hmm_qzz<<<((T_DIM - 1) * B_DIM) / 8, 256, 0, stream>>>(alpha, lbeta, logp, trans, qzz);
}
